// ComplementaryContrastiveLearning_21517786153383
// MI455X (gfx1250) — matrix operand order repaired and checked by reading, compiled, not run on hardware
//
#include <hip/hip_runtime.h>
#include <hip/hip_fp16.h>

#define Bn 8192
#define Dn 512
#define INV_T (1.0f / 0.07f)

typedef __attribute__((ext_vector_type(16))) _Float16 v16h;
typedef __attribute__((ext_vector_type(8)))  _Float16 v8h;
typedef __attribute__((ext_vector_type(4)))  _Float16 v4h;
typedef __attribute__((ext_vector_type(8)))  float    v8f;

// Wave32 XOR-butterfly add in one DS instruction:
// ds_swizzle_b32 group-of-32 mode, offset = {xor[4:0]<<10 | or=0<<5 | and=0x1f}
#define XOR_ADD_F(s, m)                                                        \
    (s) += __int_as_float(                                                     \
        __builtin_amdgcn_ds_swizzle(__float_as_int(s), 0x1f | ((m) << 10)))

// ---------------------------------------------------------------------------
// Kernel 1: L2-normalize rows of visual/textual feats, cast to f16 in ws.
// One 128-thread block per row (grid 2*Bn). Matches reference x/max(||x||,1e-12).
// ---------------------------------------------------------------------------
__global__ __launch_bounds__(128)
void ccl_normalize(const float* __restrict__ vis, const float* __restrict__ txt,
                   _Float16* __restrict__ Vn, _Float16* __restrict__ Tn) {
    int row = blockIdx.x;
    const float* src;
    _Float16*    dst;
    if (row < Bn) { src = vis + (size_t)row * Dn;        dst = Vn + (size_t)row * Dn; }
    else          { src = txt + (size_t)(row - Bn) * Dn; dst = Tn + (size_t)(row - Bn) * Dn; }

    int t = threadIdx.x;                       // 128 threads, 4 floats each
    float4 v = ((const float4*)src)[t];
    float ss = v.x * v.x + v.y * v.y + v.z * v.z + v.w * v.w;
    XOR_ADD_F(ss, 1);
    XOR_ADD_F(ss, 2);
    XOR_ADD_F(ss, 4);
    XOR_ADD_F(ss, 8);
    XOR_ADD_F(ss, 16);

    __shared__ float red[4];
    if ((t & 31) == 0) red[t >> 5] = ss;
    __syncthreads();
    float tot = red[0] + red[1] + red[2] + red[3];
    float inv = 1.0f / fmaxf(sqrtf(tot), 1e-12f);

    v4h o;
    o[0] = (_Float16)(v.x * inv);
    o[1] = (_Float16)(v.y * inv);
    o[2] = (_Float16)(v.z * inv);
    o[3] = (_Float16)(v.w * inv);
    ((v4h*)dst)[t] = o;
}

// ---------------------------------------------------------------------------
// Kernel 2: tiled WMMA GEMM  sims = Vn @ Tn^T  (f16 in, f32 acc/out), fused
// masked-exp row-partial-sum epilogue.
// Grid (64,64); block 256 threads = 8 waves; block tile 128x128.
// Waves in 2x4 arrangement -> 64x32 per wave = 4x2 v_wmma_f32_16x16x32_f16.
// Double-buffered K loop (16 steps of K=32) hides L2 load latency behind
// each 8-WMMA group.
// ---------------------------------------------------------------------------
__device__ __forceinline__ void ccl_load_frags(const _Float16* __restrict__ Vn,
                                               const _Float16* __restrict__ Tn,
                                               int rowBase, int colBase,
                                               int l15, int lhalf, int k0,
                                               v16h a[4], v16h b[2]) {
    // A fragment (16x32 f16): lane holds row (l15); K-chunks k0+8*lhalf and +16.
    #pragma unroll
    for (int mi = 0; mi < 4; ++mi) {
        const _Float16* p = Vn + (size_t)(rowBase + mi * 16 + l15) * Dn + k0 + lhalf * 8;
        v8h lo = *(const v8h*)p;
        v8h hi = *(const v8h*)(p + 16);
        a[mi] = __builtin_shufflevector(lo, hi,
                    0, 1, 2, 3, 4, 5, 6, 7, 8, 9, 10, 11, 12, 13, 14, 15);
    }
    // B fragment (32x16 f16): lane holds column (l15); K-chunks k0+8*lhalf and +16.
    #pragma unroll
    for (int ni = 0; ni < 2; ++ni) {
        const _Float16* q = Tn + (size_t)(colBase + ni * 16 + l15) * Dn
                               + k0 + lhalf * 8;
        v8h blo = *(const v8h*)q;
        v8h bhi = *(const v8h*)(q + 16);
        b[ni] = __builtin_shufflevector(blo, bhi,
                    0, 1, 2, 3, 4, 5, 6, 7, 8, 9, 10, 11, 12, 13, 14, 15);
    }
}

__device__ __forceinline__ void ccl_mma8(v8f acc[4][2], const v16h a[4], const v16h b[2]) {
    #pragma unroll
    for (int mi = 0; mi < 4; ++mi)
        #pragma unroll
        for (int ni = 0; ni < 2; ++ni)
            acc[mi][ni] = __builtin_amdgcn_wmma_f32_16x16x32_f16(
                false, a[mi], false, b[ni], (short)0, acc[mi][ni], false, false);
}

__global__ __launch_bounds__(256)
void ccl_gemm(const _Float16* __restrict__ Vn, const _Float16* __restrict__ Tn,
              const int* __restrict__ noisy,
              float* __restrict__ sims, float* __restrict__ partials) {
    const int wid   = threadIdx.x >> 5;
    const int lane  = threadIdx.x & 31;
    const int waveM = wid >> 2;             // 0..1
    const int waveN = wid & 3;              // 0..3
    const int rowBase = blockIdx.y * 128 + waveM * 64;
    const int colBase = blockIdx.x * 128 + waveN * 32;
    const int lhalf = lane >> 4;            // 0 | 1
    const int l15   = lane & 15;

    v8f acc[4][2] = {};
    v16h a0[4], b0[2], a1[4], b1[2];

    ccl_load_frags(Vn, Tn, rowBase, colBase, l15, lhalf, 0, a0, b0);
    #pragma unroll
    for (int k = 0; k < 16; k += 2) {
        if (k + 1 < 16)
            ccl_load_frags(Vn, Tn, rowBase, colBase, l15, lhalf, (k + 1) * 32, a1, b1);
        ccl_mma8(acc, a0, b0);
        if (k + 2 < 16)
            ccl_load_frags(Vn, Tn, rowBase, colBase, l15, lhalf, (k + 2) * 32, a0, b0);
        if (k + 1 < 16)
            ccl_mma8(acc, a1, b1);
    }

    // Epilogue. C/D layout: VGPR r -> lanes 0-15 M=r, lanes 16-31 M=r+8; N=l15.
    // Per-lane column noisy flags as 0/1 floats; boolean OR == fmaxf.
    const float fj0 = noisy[colBase + l15]      ? 1.0f : 0.0f;
    const float fj1 = noisy[colBase + 16 + l15] ? 1.0f : 0.0f;
    const int partCol = blockIdx.x * 4 + waveN;   // 0..255

    #pragma unroll
    for (int mi = 0; mi < 4; ++mi) {
        #pragma unroll
        for (int r = 0; r < 8; ++r) {
            int row = rowBase + mi * 16 + lhalf * 8 + r;
            float fr = noisy[row] ? 1.0f : 0.0f;
            float s0 = acc[mi][0][r];
            float s1 = acc[mi][1][r];
            sims[(size_t)row * Bn + colBase + l15]      = s0;
            sims[(size_t)row * Bn + colBase + 16 + l15] = s1;
            float rsum = __expf(s0 * INV_T) * fmaxf(fr, fj0)
                       + __expf(s1 * INV_T) * fmaxf(fr, fj1);
            // reduce across the 16 lanes of each half-group (same row)
            XOR_ADD_F(rsum, 1);
            XOR_ADD_F(rsum, 2);
            XOR_ADD_F(rsum, 4);
            XOR_ADD_F(rsum, 8);
            if (l15 == 0) partials[(size_t)row * 256 + partCol] = rsum;
        }
    }
}

// ---------------------------------------------------------------------------
// Kernel 3: negative_mask[i][j] = noisy[i] | noisy[j]  (pure store-bound).
// float2 stores keep the 8-byte alignment of d_out+2.
// ---------------------------------------------------------------------------
__global__ __launch_bounds__(256)
void ccl_mask(const int* __restrict__ noisy, float* __restrict__ mout) {
    size_t idx = (size_t)blockIdx.x * blockDim.x + threadIdx.x; // one float2
    size_t i = idx >> 12;                  // Bn/2 = 4096 float2 per row
    int j = (int)(idx & 4095) << 1;
    int ni = noisy[i];
    float2 o;
    o.x = (ni | noisy[j])     ? 1.0f : 0.0f;
    o.y = (ni | noisy[j + 1]) ? 1.0f : 0.0f;
    ((float2*)mout)[idx] = o;
}

// ---------------------------------------------------------------------------
// Kernel 4: per-row sum of 256 partials -> log(sum + eps). One wave per row.
// ---------------------------------------------------------------------------
__global__ __launch_bounds__(256)
void ccl_rowlog(const float* __restrict__ partials, float* __restrict__ logbuf) {
    int wid  = threadIdx.x >> 5;
    int lane = threadIdx.x & 31;
    int row  = blockIdx.x * 8 + wid;
    float s = 0.0f;
    #pragma unroll
    for (int c = 0; c < 8; ++c) s += partials[(size_t)row * 256 + c * 32 + lane];
    XOR_ADD_F(s, 1);
    XOR_ADD_F(s, 2);
    XOR_ADD_F(s, 4);
    XOR_ADD_F(s, 8);
    XOR_ADD_F(s, 16);
    if (lane == 0) logbuf[row] = __logf(s + 1e-8f);
}

// ---------------------------------------------------------------------------
// Kernel 5: mean over Bn logs -> both scalar losses.
// ---------------------------------------------------------------------------
__global__ __launch_bounds__(256)
void ccl_finalize(const float* __restrict__ logbuf, float* __restrict__ out) {
    float s = 0.0f;
    for (int i = threadIdx.x; i < Bn; i += 256) s += logbuf[i];
    XOR_ADD_F(s, 1);
    XOR_ADD_F(s, 2);
    XOR_ADD_F(s, 4);
    XOR_ADD_F(s, 8);
    XOR_ADD_F(s, 16);
    __shared__ float red[8];
    if ((threadIdx.x & 31) == 0) red[threadIdx.x >> 5] = s;
    __syncthreads();
    if (threadIdx.x == 0) {
        float t = 0.0f;
        #pragma unroll
        for (int i = 0; i < 8; ++i) t += red[i];
        float mean = t / (float)Bn;
        out[0] = mean;   // complementary_loss
        out[1] = mean;   // brownian_loss (identical by construction)
    }
}

// ---------------------------------------------------------------------------
extern "C" void kernel_launch(void* const* d_in, const int* in_sizes, int n_in,
                              void* d_out, int out_size, void* d_ws, size_t ws_size,
                              hipStream_t stream) {
    const float* vis   = (const float*)d_in[0];
    const float* txt   = (const float*)d_in[1];
    // d_in[2] = confident_clean_mask (unused by the reference math)
    const int*   noisy = (const int*)d_in[3];

    float* out      = (float*)d_out;
    float* mask_out = out + 2;
    float* sims_out = out + 2 + (size_t)Bn * Bn;

    // Workspace layout: Vn(8MB) | Tn(8MB) | partials(8MB) | logbuf(32KB)
    char* ws = (char*)d_ws;
    _Float16* Vn       = (_Float16*)ws;
    _Float16* Tn       = (_Float16*)(ws + (size_t)Bn * Dn * 2);
    float*    partials = (float*)(ws + (size_t)2 * Bn * Dn * 2);
    float*    logbuf   = (float*)(ws + (size_t)2 * Bn * Dn * 2 + (size_t)Bn * 256 * 4);

    ccl_normalize<<<2 * Bn, 128, 0, stream>>>(vis, txt, Vn, Tn);
    ccl_gemm<<<dim3(64, 64), 256, 0, stream>>>(Vn, Tn, noisy, sims_out, partials);
    ccl_mask<<<(unsigned)(((size_t)Bn * Bn / 2) / 256), 256, 0, stream>>>(noisy, mask_out);
    ccl_rowlog<<<Bn / 8, 256, 0, stream>>>(partials, logbuf);
    ccl_finalize<<<1, 256, 0, stream>>>(logbuf, out);
}
